// AMEGNN_13108240188113
// MI455X (gfx1250) — matrix operand order repaired and checked by reading, not compiled, not run on hardware
//
#include <hip/hip_runtime.h>
#include <cstdint>
#include <cstddef>

// ---------------------------------------------------------------------------
// Problem constants (from the reference)
// ---------------------------------------------------------------------------
#define NN      10000       // nodes
#define NE      160000      // edges
#define CCH     14          // channels
#define HDIM    128
#define RDIM    64
#define NLAY    5
#define KRR     224         // 196 padded to multiple of 32

typedef unsigned short u16;
typedef __attribute__((ext_vector_type(16))) __bf16   v16bf;
typedef __attribute__((ext_vector_type(8)))  float    v8f;
typedef __attribute__((ext_vector_type(4)))  unsigned int v4u;

union FragBF { v16bf v; unsigned int u[8]; v4u q[2]; };

__device__ __forceinline__ float bf2f(u16 h) {
    unsigned int u = ((unsigned int)h) << 16;
    return __uint_as_float(u);
}
__device__ __forceinline__ u16 f2bf(float f) {
    unsigned int u = __float_as_uint(f);
    return (u16)((u + 0x7fffu + ((u >> 16) & 1u)) >> 16);
}
// fast SiLU: v_exp_f32 + v_add + v_rcp_f32 + v_mul (avoids precise-fdiv chain)
__device__ __forceinline__ float siluf(float x) {
    return x * __builtin_amdgcn_rcpf(1.f + __expf(-x));
}

// ---- CDNA5 async global->LDS copy (ASYNCcnt-tracked, no VGPR staging) ----
__device__ __forceinline__ void async_copy_b128(unsigned lds_byte_addr, const void* gaddr) {
    asm volatile("global_load_async_to_lds_b128 %0, %1, off"
                 :: "v"(lds_byte_addr), "v"(gaddr) : "memory");
}
__device__ __forceinline__ void wait_async_zero() {
    asm volatile("s_wait_asynccnt 0" ::: "memory");
}

// ---------------------------------------------------------------------------
// Elementwise / utility kernels
// ---------------------------------------------------------------------------
__global__ void k_zero(float* p, long n) {
    long i = (long)blockIdx.x * blockDim.x + threadIdx.x;
    if (i < n) p[i] = 0.f;
}
__global__ void k_copyf(float* dst, const float* src, long n) {
    long i = (long)blockIdx.x * blockDim.x + threadIdx.x;
    if (i < n) dst[i] = src[i];
}
__global__ void k_f2bf(u16* dst, const float* src, long n) {
    long i = (long)blockIdx.x * blockDim.x + threadIdx.x;
    if (i < n) dst[i] = f2bf(src[i]);
}
__global__ void k_degree(const int* row, float* deg, int e_cnt) {
    int i = blockIdx.x * blockDim.x + threadIdx.x;
    if (i < e_cnt) atomicAdd(&deg[row[i]], 1.f);
}

// ---------------------------------------------------------------------------
// Weight packing: fp32 W[din,dout] (row major) -> bf16 WMMA B fragments.
// B-fragment lane layout (wave32, 16-bit B 32x16):
//   n = lane & 15 ; k = tk*32 + (h<8 ? h : h+8) + 8*(lane>>4)
// ---------------------------------------------------------------------------
__global__ void k_pack(const float* W, u16* dst, int din, int dout) {
    int ntk = (din + 31) >> 5;
    int ntn = (dout + 15) >> 4;
    long total = (long)ntk * ntn * 512;
    long idx = (long)blockIdx.x * blockDim.x + threadIdx.x;
    if (idx >= total) return;
    int tile   = (int)(idx >> 9);
    int within = (int)(idx & 511);
    int lane = within >> 4;
    int h    = within & 15;
    int tn = tile / ntk;
    int tk = tile - tn * ntk;
    int k = tk * 32 + ((h < 8) ? h : (h + 8)) + 8 * (lane >> 4);
    int n = tn * 16 + (lane & 15);
    float val = (k < din && n < dout) ? W[(long)k * dout + n] : 0.f;
    dst[idx] = f2bf(val);
}

// ---------------------------------------------------------------------------
// bf16 WMMA GEMM, templated on K (register-resident A fragments) and on the
// epilogue configuration (bias / SiLU / fp32-out / bf16-out) so the hot loop
// and stores are branch-free.
// Block = 128 threads (4 waves), 64 rows/block. Full blocks stage A via
// GLOBAL_LOAD_ASYNC_TO_LDS_B128 (ASYNCcnt); tail block uses guarded staging.
// A-fragment (16-bit A 16x32): m=lane&15, VGPR v: k=(v<4?2v:8+2v)+8*(lane>>4)
// D layout: VGPR v -> m = v + 8*(lane>>4), n = lane&15
// ---------------------------------------------------------------------------
template<int NTK, bool BIAS, bool ACT, bool WF, bool WB>
__global__ __launch_bounds__(128) void k_gemm_t(
    const unsigned int* Apairs,   // A as dword pairs of bf16, ld = KPAD/2
    int M,
    const u16* Bp,                // packed weights
    int Nreal, int Npad,
    const float* bias,
    float* outF, u16* outB, int ldOut)
{
    constexpr int KPAD = NTK * 32;
    constexpr int KD   = KPAD / 2;           // dwords per row
    __shared__ unsigned int As[64 * KD];

    const int tid  = threadIdx.x;
    const int wave = tid >> 5;
    const int lane = tid & 31;
    const int hi   = lane >> 4;
    const int ln   = lane & 15;
    const long rowBase = (long)blockIdx.x * 64;
    const bool fullTile = (rowBase + 64 <= (long)M);

    if (fullTile) {
        // ---- async tile staging: 64*KPAD*2 bytes, 16B per lane per round ----
        const unsigned ldsBase = (unsigned)(unsigned long long)(uintptr_t)&As[0];
        const char* gBase = (const char*)Apairs + rowBase * (KPAD * 2);
#pragma unroll
        for (int r = 0; r < NTK * 2; ++r) {          // rounds = 64*KPAD*2 / (128*16)
            unsigned boff = (unsigned)(r * 2048 + tid * 16);
            async_copy_b128(ldsBase + boff, gBase + boff);
        }
        wait_async_zero();
    } else {
        for (int idx = tid; idx < 64 * KD; idx += 128) {
            int r = idx / KD, dcol = idx - r * KD;
            long gr = rowBase + r;
            As[r * KD + dcol] = (gr < M) ? Apairs[gr * KD + dcol] : 0u;
        }
    }
    __syncthreads();

    // ---- hoist this wave's A fragments into registers (reused for all tn) ----
    const int rloc = wave * 16 + ln;
    FragBF afr[NTK];
#pragma unroll
    for (int tk = 0; tk < NTK; ++tk) {
#pragma unroll
        for (int v = 0; v < 8; ++v) {
            int k = tk * 32 + ((v < 4) ? (2 * v) : (8 + 2 * v)) + 8 * hi;
            afr[tk].u[v] = As[rloc * KD + (k >> 1)];
        }
    }

    for (int tn = 0; tn * 16 < Npad; ++tn) {
        v8f acc = {};
        const u16* btile = Bp + (long)tn * NTK * 512;
#pragma unroll
        for (int tk = 0; tk < NTK; ++tk) {
            FragBF bfm;
            const v4u* bq = (const v4u*)(btile + tk * 512 + lane * 16);
            bfm.q[0] = bq[0];
            bfm.q[1] = bq[1];
            acc = __builtin_amdgcn_wmma_f32_16x16x32_bf16(
                false, afr[tk].v, false, bfm.v, (short)0, acc, false, false);
        }
        // ---- epilogue (compile-time configured) ----
        const int  n   = tn * 16 + ln;
        const bool nok = (n < Nreal);
        float bv = 0.f;
        if (BIAS) bv = nok ? bias[n] : 0.f;
        if (fullTile) {                        // uniform fast path
            if (nok) {
#pragma unroll
                for (int v = 0; v < 8; ++v) {
                    long m = rowBase + wave * 16 + v + 8 * hi;
                    float x = acc[v] + bv;
                    if (ACT) x = siluf(x);
                    if (WF) outF[m * (long)ldOut + n] = x;
                    if (WB) outB[m * (long)ldOut + n] = f2bf(x);
                }
            }
        } else {
#pragma unroll
            for (int v = 0; v < 8; ++v) {
                long m = rowBase + wave * 16 + v + 8 * hi;
                if (m < M && nok) {
                    float x = acc[v] + bv;
                    if (ACT) x = siluf(x);
                    if (WF) outF[m * (long)ldOut + n] = x;
                    if (WB) outB[m * (long)ldOut + n] = f2bf(x);
                }
            }
        }
    }
}

// ---------------------------------------------------------------------------
// Edge geometry: cd, rr_raw (cd . cd^T), ra_raw closed form, fp32 norms.
//  ra[i,j] = sum_c (a_i - b_j)_c * (C*a_i - sum_k b_k)_c
// Outputs bf16 rows zero-padded to KRR=224.
// ---------------------------------------------------------------------------
__global__ void k_geom(const float* coordW, const int* row, const int* col,
                       float* cd_out, u16* rr_out, u16* ra_out,
                       float* nrr, float* nra)
{
    int e = blockIdx.x * blockDim.x + threadIdx.x;
    if (e >= NE) return;
    float a[CCH][3], d[CCH][3];
    float sb0 = 0.f, sb1 = 0.f, sb2 = 0.f;
    const float* pa = coordW + (long)row[e] * (CCH * 3);
    const float* pb = coordW + (long)col[e] * (CCH * 3);
    float* pcd = cd_out + (long)e * (CCH * 3);
#pragma unroll
    for (int i = 0; i < CCH; ++i) {
#pragma unroll
        for (int k = 0; k < 3; ++k) {
            float av = pa[i * 3 + k], bv = pb[i * 3 + k];
            a[i][k] = av; d[i][k] = av - bv;
            pcd[i * 3 + k] = av - bv;
        }
        sb0 += a[i][0] - d[i][0];
        sb1 += a[i][1] - d[i][1];
        sb2 += a[i][2] - d[i][2];
    }
    long rbase = (long)e * KRR;
    float s2rr = 0.f, s2ra = 0.f;
#pragma unroll
    for (int i = 0; i < CCH; ++i) {
        float s0 = (float)CCH * a[i][0] - sb0;
        float s1 = (float)CCH * a[i][1] - sb1;
        float s2 = (float)CCH * a[i][2] - sb2;
#pragma unroll
        for (int j = 0; j < CCH; ++j) {
            float drr = d[i][0] * d[j][0] + d[i][1] * d[j][1] + d[i][2] * d[j][2];
            float b0 = a[j][0] - d[j][0];
            float b1 = a[j][1] - d[j][1];
            float b2 = a[j][2] - d[j][2];
            float dra = (a[i][0] - b0) * s0 + (a[i][1] - b1) * s1 + (a[i][2] - b2) * s2;
            rr_out[rbase + i * CCH + j] = f2bf(drr);
            ra_out[rbase + i * CCH + j] = f2bf(dra);
            s2rr += drr * drr;
            s2ra += dra * dra;
        }
    }
#pragma unroll
    for (int p = CCH * CCH; p < KRR; ++p) { rr_out[rbase + p] = 0; ra_out[rbase + p] = 0; }
    nrr[e] = sqrtf(s2rr);
    nra[e] = sqrtf(s2ra);
}

// radial = 0.25*rr/(|rr_raw|+1) + 0.75*ra/(|ra_raw|+1) -> efin cols 256..319
// (packed 2 x bf16 per 32-bit store; efin viewed as uint rows of 160)
__global__ void k_combine(const float* rrF, const float* raF,
                          const float* nrr, const float* nra, unsigned* efinu)
{
    long idx = (long)blockIdx.x * blockDim.x + threadIdx.x;
    if (idx >= (long)NE * 32) return;
    long e = idx >> 5; int p = (int)(idx & 31); int j = 2 * p;
    float irr = __builtin_amdgcn_rcpf(nrr[e] + 1.f);
    float ira = __builtin_amdgcn_rcpf(nra[e] + 1.f);
    float v0 = 0.25f * rrF[e * RDIM + j]     * irr + 0.75f * raF[e * RDIM + j]     * ira;
    float v1 = 0.25f * rrF[e * RDIM + j + 1] * irr + 0.75f * raF[e * RDIM + j + 1] * ira;
    efinu[e * 160 + 128 + p] = (unsigned)f2bf(v0) | ((unsigned)f2bf(v1) << 16);
}

// efin cols 0..127 = h[row], 128..255 = h[col]  (32-bit packed copies)
__global__ void k_gather_h(const unsigned* hBu, const int* row, const int* col,
                           unsigned* efinu)
{
    long idx = (long)blockIdx.x * blockDim.x + threadIdx.x;
    if (idx >= (long)NE * 64) return;
    long e = idx >> 6; int c = (int)(idx & 63);
    efinu[e * 160 + c]      = hBu[(long)row[e] * 64 + c];
    efinu[e * 160 + 64 + c] = hBu[(long)col[e] * 64 + c];
}

// trans = cd * cw ; scatter into num[node, C, 3]
__global__ void k_trans_scatter(const float* cd, const float* cwF,
                                const int* row, float* num)
{
    long idx = (long)blockIdx.x * blockDim.x + threadIdx.x;
    if (idx >= (long)NE * (CCH * 3)) return;
    long e = idx / (CCH * 3);
    int t = (int)(idx - e * (CCH * 3));
    int ch = t / 3;
    float v = cd[idx] * cwF[e * CCH + ch];
    atomicAdd(&num[(long)row[e] * (CCH * 3) + t], v);
}

// coord += num / max(deg,1); re-zero num for next layer
__global__ void k_coord_update(float* coordW, float* num, const float* deg)
{
    long idx = (long)blockIdx.x * blockDim.x + threadIdx.x;
    if (idx >= (long)NN * (CCH * 3)) return;
    long n = idx / (CCH * 3);
    float inv = __builtin_amdgcn_rcpf(fmaxf(deg[n], 1.f));
    coordW[idx] += num[idx] * inv;
    num[idx] = 0.f;
}

// agg[node] += ef[edge]  (fp32 atomics for reference-grade accumulation)
__global__ void k_agg_scatter(const u16* efB, const int* row, float* agg)
{
    long idx = (long)blockIdx.x * blockDim.x + threadIdx.x;
    if (idx >= (long)NE * HDIM) return;
    long e = idx >> 7; int c = (int)(idx & 127);
    atomicAdd(&agg[(long)row[e] * HDIM + c], bf2f(efB[idx]));
}

// nodein = [hB, bf16(agg)]; re-zero agg for next layer
__global__ void k_node_gather(const u16* hB, float* agg, u16* nodein)
{
    long idx = (long)blockIdx.x * blockDim.x + threadIdx.x;
    if (idx >= (long)NN * HDIM) return;
    long n = idx >> 7; int c = (int)(idx & 127);
    nodein[n * 256 + c]       = hB[idx];
    nodein[n * 256 + 128 + c] = f2bf(agg[idx]);
    agg[idx] = 0.f;
}

// h += m (residual); refresh bf16 mirror
__global__ void k_residual(float* hF, u16* hB, const float* mF)
{
    long idx = (long)blockIdx.x * blockDim.x + threadIdx.x;
    if (idx >= (long)NN * HDIM) return;
    float v = hF[idx] + mF[idx];
    hF[idx] = v;
    hB[idx] = f2bf(v);
}

// ---------------------------------------------------------------------------
// Host orchestration
// ---------------------------------------------------------------------------
static inline int nblk(long n, int t) { return (int)((n + t - 1) / t); }
static inline size_t packBytes(int din, int dout) {
    return (size_t)((din + 31) / 32) * ((dout + 15) / 16) * 512 * sizeof(u16);
}

// dispatch over the 4 epilogue combos actually used
template<int NTK>
static void launch_g(hipStream_t s, const unsigned int* A, int M,
                     const u16* Bp, int Nreal, int Npad,
                     const float* bias, int act, float* outF, u16* outB, int ldOut)
{
    dim3 g((M + 63) / 64);
    if (bias && act && outB && !outF)
        k_gemm_t<NTK, true,  true,  false, true ><<<g, 128, 0, s>>>(A, M, Bp, Nreal, Npad, bias, outF, outB, ldOut);
    else if (bias && !act && outF && outB)
        k_gemm_t<NTK, true,  false, true,  true ><<<g, 128, 0, s>>>(A, M, Bp, Nreal, Npad, bias, outF, outB, ldOut);
    else if (bias && !act && outF)
        k_gemm_t<NTK, true,  false, true,  false><<<g, 128, 0, s>>>(A, M, Bp, Nreal, Npad, bias, outF, outB, ldOut);
    else
        k_gemm_t<NTK, false, false, true,  false><<<g, 128, 0, s>>>(A, M, Bp, Nreal, Npad, bias, outF, outB, ldOut);
}

struct Launcher {
    hipStream_t s;
    void gemm(const void* Abf, int M, int Kpad, const u16* Bp, int Nreal, int Npad,
              const float* bias, int act, float* outF, u16* outB, int ldOut) {
        const unsigned int* A = (const unsigned int*)Abf;
        switch (Kpad) {
        case 128: launch_g<4>(s, A, M, Bp, Nreal, Npad, bias, act, outF, outB, ldOut); break;
        case 224: launch_g<7>(s, A, M, Bp, Nreal, Npad, bias, act, outF, outB, ldOut); break;
        case 256: launch_g<8>(s, A, M, Bp, Nreal, Npad, bias, act, outF, outB, ldOut); break;
        case 320: launch_g<10>(s, A, M, Bp, Nreal, Npad, bias, act, outF, outB, ldOut); break;
        default: break;
        }
    }
    void pack(const float* W, u16* dst, int din, int dout) {
        long total = (long)((din + 31) / 32) * ((dout + 15) / 16) * 512;
        k_pack<<<nblk(total, 256), 256, 0, s>>>(W, dst, din, dout);
    }
};

extern "C" void kernel_launch(void* const* d_in, const int* in_sizes, int n_in,
                              void* d_out, int out_size, void* d_ws, size_t ws_size,
                              hipStream_t stream)
{
    (void)in_sizes; (void)n_in; (void)out_size; (void)ws_size;
    const float* h_in     = (const float*)d_in[0];
    const float* coord_in = (const float*)d_in[1];
    const int*   row      = (const int*)d_in[2];
    const int*   col      = (const int*)d_in[3];
    // d_in[4] channel_attr, d_in[5] channel_weights: unused by reference math
    float* out = (float*)d_out;

    // Param leaves (JAX pytree order: sorted dict keys, None dropped):
    // layers[0..4] then lin_in{b,w} then lin_out{b,w}.
    // Per layer (coord, edge, node, rl1, rl2):
    //  +0 coord.b0  +1 coord.w0(128x128) +2 coord.w1(128x14)
    //  +3 edge.b0   +4 edge.w0(320x128)  +5 edge.b1  +6 edge.w1(128x128)
    //  +7 node.b0   +8 node.w0(256x128)  +9 node.b1  +10 node.w1(128x128)
    //  +11 rl1.b0 +12 rl1.w0(196x256) +13 rl1.b1 +14 rl1.w1(256x128) +15 rl1.b2 +16 rl1.w2(128x64)
    //  +17..22 rl2 likewise
    const int PBASE = 6;
    const int LIN_IN_B  = PBASE + NLAY * 23 + 0;
    const int LIN_IN_W  = PBASE + NLAY * 23 + 1;
    const int LIN_OUT_B = PBASE + NLAY * 23 + 2;
    const int LIN_OUT_W = PBASE + NLAY * 23 + 3;

    // ---- workspace bump allocator (256B aligned) ----
    char* ws = (char*)d_ws;
    size_t off = 0;
    auto alloc = [&](size_t bytes) -> char* {
        char* p = ws + off;
        off = (off + bytes + 255) & ~(size_t)255;
        return p;
    };

    float* coordW = (float*)alloc((size_t)NN * CCH * 3 * 4);
    float* deg    = (float*)alloc((size_t)NN * 4);
    float* num    = (float*)alloc((size_t)NN * CCH * 3 * 4);
    float* agg    = (float*)alloc((size_t)NN * HDIM * 4);
    float* hF     = (float*)alloc((size_t)NN * HDIM * 4);
    u16*   hB     = (u16*)  alloc((size_t)NN * HDIM * 2);
    u16*   hbf0   = (u16*)  alloc((size_t)NN * HDIM * 2);
    u16*   nodein = (u16*)  alloc((size_t)NN * 256 * 2);
    u16*   n128   = (u16*)  alloc((size_t)NN * HDIM * 2);
    float* mF     = (float*)alloc((size_t)NN * HDIM * 4);

    float* cdbuf  = (float*)alloc((size_t)NE * CCH * 3 * 4);
    u16*   rrbuf  = (u16*)  alloc((size_t)NE * KRR * 2);
    u16*   rabuf  = (u16*)  alloc((size_t)NE * KRR * 2);
    float* nrr    = (float*)alloc((size_t)NE * 4);
    float* nra    = (float*)alloc((size_t)NE * 4);
    u16*   t256   = (u16*)  alloc((size_t)NE * 256 * 2);
    u16*   t128   = (u16*)  alloc((size_t)NE * 128 * 2);
    float* rrF    = (float*)alloc((size_t)NE * RDIM * 4);
    float* raF    = (float*)alloc((size_t)NE * RDIM * 4);
    u16*   efin   = (u16*)  alloc((size_t)NE * 320 * 2);
    u16*   e128a  = (u16*)  alloc((size_t)NE * 128 * 2);
    u16*   efB    = (u16*)  alloc((size_t)NE * 128 * 2);
    u16*   c128   = (u16*)  alloc((size_t)NE * 128 * 2);
    float* cwF    = (float*)alloc((size_t)NE * CCH * 4);

    // packed weights: per layer {c0,c1,e0,e1,n0,n1,r10,r11,r12,r20,r21,r22}
    static const int pdin[12]  = {128, 128, 320, 128, 256, 128, 196, 256, 128, 196, 256, 128};
    static const int pdout[12] = {128,  14, 128, 128, 128, 128, 256, 128,  64, 256, 128,  64};
    static const int pwleaf[12]= {  1,   2,   4,   6,   8,  10,  12,  14,  16,  18,  20,  22};
    u16* P[NLAY][12];
    for (int l = 0; l < NLAY; ++l)
        for (int j = 0; j < 12; ++j)
            P[l][j] = (u16*)alloc(packBytes(pdin[j], pdout[j]));
    u16* Pin  = (u16*)alloc(packBytes(128, 128));
    u16* Pout = (u16*)alloc(packBytes(128, 128));

    Launcher L{stream};

    // ---- init ----
    k_zero<<<nblk(NN, 256), 256, 0, stream>>>(deg, NN);
    k_zero<<<nblk((long)NN * CCH * 3, 256), 256, 0, stream>>>(num, (long)NN * CCH * 3);
    k_zero<<<nblk((long)NN * HDIM, 256), 256, 0, stream>>>(agg, (long)NN * HDIM);
    k_degree<<<nblk(NE, 256), 256, 0, stream>>>(row, deg, NE);
    k_copyf<<<nblk((long)NN * CCH * 3, 256), 256, 0, stream>>>(coordW, coord_in, (long)NN * CCH * 3);
    k_f2bf<<<nblk((long)NN * HDIM, 256), 256, 0, stream>>>(hbf0, h_in, (long)NN * HDIM);

    // ---- pack all weights ----
    for (int l = 0; l < NLAY; ++l) {
        int base = PBASE + l * 23;
        for (int j = 0; j < 12; ++j)
            L.pack((const float*)d_in[base + pwleaf[j]], P[l][j], pdin[j], pdout[j]);
    }
    L.pack((const float*)d_in[LIN_IN_W],  Pin,  128, 128);
    L.pack((const float*)d_in[LIN_OUT_W], Pout, 128, 128);

    // ---- lin_in: h = h_in @ W + b ----
    L.gemm(hbf0, NN, 128, Pin, 128, 128, (const float*)d_in[LIN_IN_B], 0, hF, hB, 128);

    // ---- layers ----
    for (int l = 0; l < NLAY; ++l) {
        int base = PBASE + l * 23;
        const float* bC0 = (const float*)d_in[base + 0];
        const float* bE0 = (const float*)d_in[base + 3];
        const float* bE1 = (const float*)d_in[base + 5];
        const float* bN0 = (const float*)d_in[base + 7];
        const float* bN1 = (const float*)d_in[base + 9];
        const float* bR10 = (const float*)d_in[base + 11];
        const float* bR11 = (const float*)d_in[base + 13];
        const float* bR12 = (const float*)d_in[base + 15];
        const float* bR20 = (const float*)d_in[base + 17];
        const float* bR21 = (const float*)d_in[base + 19];
        const float* bR22 = (const float*)d_in[base + 21];

        // geometry
        k_geom<<<nblk(NE, 256), 256, 0, stream>>>(coordW, row, col,
                                                  cdbuf, rrbuf, rabuf, nrr, nra);
        // radial MLP 1 (residue-level)
        L.gemm(rrbuf, NE, KRR, P[l][6], 256, 256, bR10, 1, nullptr, t256, 256);
        L.gemm(t256,  NE, 256, P[l][7], 128, 128, bR11, 1, nullptr, t128, 128);
        L.gemm(t128,  NE, 128, P[l][8],  64,  64, bR12, 0, rrF, nullptr, 64);
        // radial MLP 2 (atom-level)
        L.gemm(rabuf, NE, KRR, P[l][9], 256, 256, bR20, 1, nullptr, t256, 256);
        L.gemm(t256,  NE, 256, P[l][10],128, 128, bR21, 1, nullptr, t128, 128);
        L.gemm(t128,  NE, 128, P[l][11], 64,  64, bR22, 0, raF, nullptr, 64);
        // radial combine + h gathers -> edge-MLP input [E,320]
        k_combine<<<nblk((long)NE * 32, 256), 256, 0, stream>>>(rrF, raF, nrr, nra, (unsigned*)efin);
        k_gather_h<<<nblk((long)NE * 64, 256), 256, 0, stream>>>((const unsigned*)hB, row, col, (unsigned*)efin);
        // edge MLP
        L.gemm(efin,  NE, 320, P[l][2], 128, 128, bE0, 1, nullptr, e128a, 128);
        L.gemm(e128a, NE, 128, P[l][3], 128, 128, bE1, 1, nullptr, efB, 128);
        // coord MLP -> per-channel weights cw [E,14]
        L.gemm(efB,  NE, 128, P[l][0], 128, 128, bC0, 1, nullptr, c128, 128);
        L.gemm(c128, NE, 128, P[l][1],  14,  16, nullptr, 0, cwF, nullptr, CCH);
        // coordinate update (mean aggregation == sum / degree)
        k_trans_scatter<<<nblk((long)NE * CCH * 3, 256), 256, 0, stream>>>(cdbuf, cwF, row, num);
        k_coord_update<<<nblk((long)NN * CCH * 3, 256), 256, 0, stream>>>(coordW, num, deg);
        // node update
        k_agg_scatter<<<nblk((long)NE * HDIM, 256), 256, 0, stream>>>(efB, row, agg);
        k_node_gather<<<nblk((long)NN * HDIM, 256), 256, 0, stream>>>(hB, agg, nodein);
        L.gemm(nodein, NN, 256, P[l][4], 128, 128, bN0, 1, nullptr, n128, 128);
        L.gemm(n128,   NN, 128, P[l][5], 128, 128, bN1, 0, mF, nullptr, 128);
        k_residual<<<nblk((long)NN * HDIM, 256), 256, 0, stream>>>(hF, hB, mF);
    }

    // ---- lin_out -> d_out[0 : N*H], coords -> d_out[N*H : N*H + N*C*3] ----
    L.gemm(hB, NN, 128, Pout, 128, 128, (const float*)d_in[LIN_OUT_B], 0, out, nullptr, 128);
    k_copyf<<<nblk((long)NN * CCH * 3, 256), 256, 0, stream>>>(
        out + (long)NN * HDIM, coordW, (long)NN * CCH * 3);
}
